// RetinaNet_52982716564152
// MI455X (gfx1250) — matrix operand order repaired and checked by reading, not compiled, not run on hardware
//
#include <hip/hip_runtime.h>
#include <hip/hip_bf16.h>

typedef __attribute__((ext_vector_type(16))) _Float16 v16h;
typedef __attribute__((ext_vector_type(8)))  _Float16 v8h;
typedef __attribute__((ext_vector_type(8)))  float    v8f;

#define A_ANCH 9
#define NCLS 80
#define SCORE_T 0.05f
#define NMS_T 0.5f
#define DPI 100
#define CIN 256
#define BATCH 4
#define BBOX_CLIP 4.135166556742356f   // log(1000/16)

__device__ __forceinline__ v8f wmma_f16(v16h a, v16h b, v8f c) {
  // D = A(16x32 f16) * B(32x16 f16) + C(16x16 f32)
  return __builtin_amdgcn_wmma_f32_16x16x32_f16(false, a, false, b, (short)0, c,
                                                false, false);
}

// two 16-byte loads -> one 16-half fragment; `gap` = element distance between
// the two 8-half groups (16 for the A weight layout, also 16 for the B pixel data)
__device__ __forceinline__ v16h load_frag(const _Float16* p, int gap) {
  v8h lo = *(const v8h*)(p);
  v8h hi = *(const v8h*)(p + gap);
  v16h r;
  #pragma unroll
  for (int i = 0; i < 8; ++i) { r[i] = lo[i]; r[i + 8] = hi[i]; }
  return r;
}

// ---------------------------------------------------------------------------
// Implicit-GEMM 3x3 SAME conv, stride 1, Cin=256.
//   xin : f16 NHWC [B][H][W][256]
//   w16 : f16 [Cout][9 taps][256]  (tap = ky*3+kx)
//   zbuf: >=512 f16 of zeros (halo source for out-of-image pixels)
//   yh  : f16 NHWC output (tower path, bias+ReLU), or nullptr
//   yf  : f32 NHWC output (head path, bias only),  or nullptr
// Wave computes a 32(cout) x 32(pixel) tile: 2x2 of 16x16 WMMA tiles.
// Block = 256 threads = 8 waves arranged 4(M) x 2(N) => 128 couts x 64 px.
// Inner loop is branchless: A rows are clamped (over-range M-tiles are simply
// not stored), out-of-image B pixels read from the zero buffer.
// ---------------------------------------------------------------------------
__global__ __launch_bounds__(256)
void conv3x3_wmma(const _Float16* __restrict__ xin,
                  const _Float16* __restrict__ w16,
                  const float*    __restrict__ bias,
                  const _Float16* __restrict__ zbuf,
                  _Float16*       __restrict__ yh,
                  float*          __restrict__ yf,
                  int H, int W, int Cout, int relu)
{
  const int HW   = H * W;
  const int b    = blockIdx.z;
  const int tid  = threadIdx.x;
  const int lane = tid & 31;
  const int wv   = tid >> 5;
  const int mBase = blockIdx.y * 128 + (wv & 3) * 32;
  const int nBase = blockIdx.x * 64  + (wv >> 2) * 32;
  if (mBase >= Cout) return;            // wave-uniform: EXEC stays all-ones

  const int lidx    = lane & 15;
  const int halfSel = lane >> 4;
  const int offA    = halfSel * 8;      // A-layout half-wave K offset

  const _Float16* xb = xin + (size_t)b * HW * CIN;

  // the two output pixels this lane contributes B-columns for
  const int p0 = nBase + lidx;
  const int p1 = p0 + 16;
  const int y0 = p0 / W, x0c = p0 % W;
  const int y1 = p1 / W, x1c = p1 % W;

  // the two weight rows this lane supplies A-data for (clamped, not guarded)
  const int rowA0 = min(mBase + lidx,      Cout - 1);
  const int rowA1 = min(mBase + lidx + 16, Cout - 1);

  v8f acc00 = {}; v8f acc01 = {}; v8f acc10 = {}; v8f acc11 = {};

  #pragma unroll 1
  for (int tap = 0; tap < 9; ++tap) {
    const int dy = tap / 3 - 1, dx = tap % 3 - 1;
    const int yy0 = y0 + dy, xx0 = x0c + dx;
    const int yy1 = y1 + dy, xx1 = x1c + dx;
    const bool v0 = (yy0 >= 0) & (yy0 < H) & (xx0 >= 0) & (xx0 < W);
    const bool v1 = (yy1 >= 0) & (yy1 < H) & (xx1 >= 0) & (xx1 < W);
    const _Float16* s0 =
        v0 ? xb + ((size_t)(yy0 * W + xx0)) * CIN + halfSel * 8 : zbuf;
    const _Float16* s1 =
        v1 ? xb + ((size_t)(yy1 * W + xx1)) * CIN + halfSel * 8 : zbuf;
    const _Float16* wr0 = w16 + ((size_t)rowA0 * 9 + tap) * CIN + offA;
    const _Float16* wr1 = w16 + ((size_t)rowA1 * 9 + tap) * CIN + offA;
    if (tap < 8) __builtin_prefetch(wr0 + CIN, 0, 1);   // global_prefetch next tap

    #pragma unroll
    for (int cb = 0; cb < 8; ++cb) {
      const int c0 = cb * 32;
      // A fragments (16-bit A layout: halves 0..7 -> K=off.., 8..15 -> K=off+16)
      v16h af0 = load_frag(wr0 + c0, 16);
      v16h af1 = load_frag(wr1 + c0, 16);
      // B fragments: lane = column(pixel), same K layout as A (off.., off+16..)
      v16h bf0 = load_frag(s0 + c0, 16);
      v16h bf1 = load_frag(s1 + c0, 16);

      acc00 = wmma_f16(af0, bf0, acc00);
      acc01 = wmma_f16(af0, bf1, acc01);
      acc10 = wmma_f16(af1, bf0, acc10);
      acc11 = wmma_f16(af1, bf1, acc11);
    }
  }

  // ---- store: C/D layout: vgpr r, lanes0-15 -> M=r, lanes16-31 -> M=r+8
  #pragma unroll
  for (int mt = 0; mt < 2; ++mt) {
    if (mBase + mt * 16 >= Cout) continue;
    const int mB = mBase + mt * 16 + offA;   // 8 consecutive couts for this lane
    #pragma unroll
    for (int nt = 0; nt < 2; ++nt) {
      v8f acc = (mt == 0) ? (nt == 0 ? acc00 : acc01)
                          : (nt == 0 ? acc10 : acc11);
      const int p = nBase + nt * 16 + lidx;
      if (yh) {
        v8h o;
        #pragma unroll
        for (int r = 0; r < 8; ++r) {
          float v = acc[r] + bias[mB + r];
          if (relu) v = fmaxf(v, 0.f);
          o[r] = (_Float16)v;
        }
        *(v8h*)(yh + (size_t)b * HW * CIN + (size_t)p * CIN + mB) = o;
      } else {
        float* dst = yf + (size_t)b * HW * Cout + (size_t)p * Cout + mB;
        #pragma unroll
        for (int r = 0; r < 8; ++r) {
          const int m = mB + r;
          if (m < Cout) dst[r] = acc[r] + bias[m];
        }
      }
    }
  }
}

// ---------------------------------------------------------------------------
// zero-fill a small f16 region (halo source)
__global__ void zero_f16(_Float16* __restrict__ z, int n)
{
  int i = blockIdx.x * 256 + threadIdx.x;
  if (i < n) z[i] = (_Float16)0.f;
}

// fp32 NCHW -> f16 NHWC
__global__ void nchw_to_nhwc_f16(const float* __restrict__ in,
                                 _Float16* __restrict__ out,
                                 int C, int HW, int total)
{
  int i = blockIdx.x * 256 + threadIdx.x;
  if (i >= total) return;
  int p = i % HW;
  int t = i / HW;
  int c = t % C;
  int bb = t / C;
  out[((size_t)bb * HW + p) * C + c] = (_Float16)in[i];
}

// fp32 [rows][256][3][3] OIHW -> f16 [rows][9][256]
__global__ void weights_to_f16(const float* __restrict__ in,
                               _Float16* __restrict__ out, int rows)
{
  int total = rows * 2304;
  int i = blockIdx.x * 256 + threadIdx.x;
  if (i >= total) return;
  int row = i / 2304;
  int rem = i % 2304;
  int ci  = rem / 9;
  int t   = rem % 9;
  out[((size_t)row * 9 + t) * 256 + ci] = (_Float16)in[i];
}

// in-place sigmoid + score-threshold mask
__global__ void sigmoid_mask(float* __restrict__ d, size_t total)
{
  size_t i = (size_t)blockIdx.x * 256 + threadIdx.x;
  size_t stride = (size_t)gridDim.x * 256;
  for (; i < total; i += stride) {
    float s = 1.f / (1.f + expf(-d[i]));
    d[i] = (s > SCORE_T) ? s : -1.f;
  }
}

// per (level, batch) top-100 via iterated block argmax (mutates masked buffer)
__global__ __launch_bounds__(256)
void topk100(float* __restrict__ cls, float* __restrict__ topS, int* __restrict__ topI)
{
  const int HWs[4] = {4096, 1024, 256, 64};
  const int lvl = blockIdx.x;
  const int b   = blockIdx.y;
  size_t start = 0;
  for (int l = 0; l < lvl; ++l) start += (size_t)BATCH * HWs[l] * 720;
  const int HW = HWs[lvl];
  float* d = cls + start + (size_t)b * HW * 720;
  const int N = HW * 720;
  __shared__ float sv[256];
  __shared__ int   si[256];
  const int tid = threadIdx.x;
  for (int k = 0; k < DPI; ++k) {
    float best = -1e30f; int bi = 0;
    for (int i = tid; i < N; i += 256) {
      float v = d[i];
      if (v > best) { best = v; bi = i; }
    }
    sv[tid] = best; si[tid] = bi;
    __syncthreads();
    for (int s = 128; s > 0; s >>= 1) {
      if (tid < s && sv[tid + s] > sv[tid]) { sv[tid] = sv[tid + s]; si[tid] = si[tid + s]; }
      __syncthreads();
    }
    if (tid == 0) {
      topS[((size_t)b * 4 + lvl) * DPI + k] = sv[0];
      topI[((size_t)b * 4 + lvl) * DPI + k] = si[0];
      d[si[0]] = -2.f;
    }
    __syncthreads();
  }
}

// anchor generation + box decode + clip
__global__ void decode_boxes(const float* __restrict__ topS, const int* __restrict__ topI,
                             const float* __restrict__ regBuf,
                             const int* __restrict__ imgHp, const int* __restrict__ imgWp,
                             float* __restrict__ candB, float* __restrict__ candS,
                             int* __restrict__ candL)
{
  const int i = blockIdx.x * 256 + threadIdx.x;
  if (i >= BATCH * 4 * DPI) return;
  const int HWs[4]     = {4096, 1024, 256, 64};
  const int Wl[4]      = {64, 32, 16, 8};
  const int strides[4] = {8, 16, 32, 64};
  const int bases[4]   = {32, 64, 128, 256};
  int b   = i / (4 * DPI);
  int rem = i % (4 * DPI);
  int lvl = rem / DPI;
  int j   = rem % DPI;
  float imgH = (float)imgHp[0], imgW = (float)imgWp[0];
  float s   = topS[((size_t)b * 4 + lvl) * DPI + j];
  int   idx = topI[((size_t)b * 4 + lvl) * DPI + j];
  int lbl  = idx % NCLS;
  int aidx = idx / NCLS;
  int p = aidx / A_ANCH;
  int a = aidx % A_ANCH;
  int W = Wl[lvl];
  int y = p / W, x = p % W;
  // anchor for (level, a)
  int s0i = bases[lvl];
  float sizes[3] = { (float)s0i,
                     (float)(int)(s0i * 1.2599210498948732),
                     (float)(int)(s0i * 1.5874010519681994) };
  const float ratios[3] = {0.5f, 1.f, 2.f};
  int ridx = a / 3, sidx = a % 3;
  float hr = sqrtf(ratios[ridx]);
  float wr = 1.f / hr;
  float wsz = wr * sizes[sidx], hsz = hr * sizes[sidx];
  float bx0 = rintf(-wsz * 0.5f), by0 = rintf(-hsz * 0.5f);
  float bx1 = rintf( wsz * 0.5f), by1 = rintf( hsz * 0.5f);
  float sx = (float)(x * strides[lvl]);
  float sy = (float)(y * strides[lvl]);
  float a0 = fminf(fmaxf(sx + bx0, 0.f), imgW);
  float a1 = fminf(fmaxf(sy + by0, 0.f), imgH);
  float a2 = fminf(fmaxf(sx + bx1, 0.f), imgW);
  float a3 = fminf(fmaxf(sy + by1, 0.f), imgH);
  // regression deltas (NHWC head layout == reference flatten order)
  size_t rstart = 0;
  for (int l = 0; l < lvl; ++l) rstart += (size_t)BATCH * HWs[l] * 36;
  const float* rb = regBuf + rstart + (size_t)b * HWs[lvl] * 36 + (size_t)p * 36 + a * 4;
  float dxv = rb[0], dyv = rb[1];
  float dwv = fminf(rb[2], BBOX_CLIP), dhv = fminf(rb[3], BBOX_CLIP);
  float w = a2 - a0, h = a3 - a1;
  float cx = a0 + 0.5f * w, cy = a1 + 0.5f * h;
  float pcx = dxv * w + cx, pcy = dyv * h + cy;
  float pw = expf(dwv) * w, ph = expf(dhv) * h;
  float q0 = fminf(fmaxf(pcx - 0.5f * pw, 0.f), imgW);
  float q1 = fminf(fmaxf(pcy - 0.5f * ph, 0.f), imgH);
  float q2 = fminf(fmaxf(pcx + 0.5f * pw, 0.f), imgW);
  float q3 = fminf(fmaxf(pcy + 0.5f * ph, 0.f), imgH);
  bool valid = s > SCORE_T;
  int o = b * 400 + lvl * DPI + j;
  candB[o * 4 + 0] = q0; candB[o * 4 + 1] = q1;
  candB[o * 4 + 2] = q2; candB[o * 4 + 3] = q3;
  candS[o] = valid ? s : -1.f;
  candL[o] = valid ? lbl : -1;
}

// per-batch bitonic sort (descending score) of the 400 candidates
__global__ __launch_bounds__(256)
void sort_cands(const float* __restrict__ candB, const float* __restrict__ candS,
                const int* __restrict__ candL,
                float* __restrict__ sortB, float* __restrict__ sortS,
                int* __restrict__ sortL)
{
  const int b = blockIdx.x;
  __shared__ float key[512];
  __shared__ int   idx[512];
  int tid = threadIdx.x;
  for (int t = tid; t < 512; t += 256) {
    if (t < 400) { key[t] = -candS[b * 400 + t]; idx[t] = t; }
    else         { key[t] = 1e30f;               idx[t] = 0; }
  }
  __syncthreads();
  for (int k = 2; k <= 512; k <<= 1)
    for (int j = k >> 1; j > 0; j >>= 1) {
      for (int t = tid; t < 512; t += 256) {
        int ixj = t ^ j;
        if (ixj > t) {
          bool up = ((t & k) == 0);
          if ((key[t] > key[ixj]) == up) {
            float kk = key[t]; key[t] = key[ixj]; key[ixj] = kk;
            int   ii = idx[t]; idx[t] = idx[ixj]; idx[ixj] = ii;
          }
        }
      }
      __syncthreads();
    }
  for (int t = tid; t < 400; t += 256) {
    int src = idx[t];
    sortS[b * 400 + t] = candS[b * 400 + src];
    sortL[b * 400 + t] = candL[b * 400 + src];
    for (int c = 0; c < 4; ++c)
      sortB[(b * 400 + t) * 4 + c] = candB[(b * 400 + src) * 4 + c];
  }
}

// class-offset NMS + final top-100 emission
__global__ __launch_bounds__(256)
void nms_final(const float* __restrict__ sortB, const float* __restrict__ sortS,
               const int* __restrict__ sortL,
               const int* __restrict__ imgHp, const int* __restrict__ imgWp,
               float* __restrict__ outB, float* __restrict__ outS,
               float* __restrict__ outL)
{
  const int b = blockIdx.x;
  __shared__ float x1[400], y1[400], x2[400], y2[400], sc[400], off[400], area[400];
  __shared__ int   lb[400], kp[400];
  int tid = threadIdx.x;
  int ih = imgHp[0], iw = imgWp[0];
  float mx = (float)(ih > iw ? ih : iw) + 1.f;
  for (int t = tid; t < 400; t += 256) {
    x1[t] = sortB[(b * 400 + t) * 4 + 0];
    y1[t] = sortB[(b * 400 + t) * 4 + 1];
    x2[t] = sortB[(b * 400 + t) * 4 + 2];
    y2[t] = sortB[(b * 400 + t) * 4 + 3];
    sc[t] = sortS[b * 400 + t];
    lb[t] = sortL[b * 400 + t];
    off[t]  = (float)lb[t] * mx;
    area[t] = (x2[t] - x1[t]) * (y2[t] - y1[t]);
    kp[t] = sc[t] > SCORE_T ? 1 : 0;
  }
  __syncthreads();
  for (int i = 0; i < 400; ++i) {
    if (kp[i]) {
      float ax1 = x1[i] + off[i], ay1 = y1[i] + off[i];
      float ax2 = x2[i] + off[i], ay2 = y2[i] + off[i];
      for (int t = tid; t < 400; t += 256) {
        if (t > i && kp[t]) {
          float bx1 = x1[t] + off[t], by1 = y1[t] + off[t];
          float bx2 = x2[t] + off[t], by2 = y2[t] + off[t];
          float lx = fmaxf(ax1, bx1), ly = fmaxf(ay1, by1);
          float rx = fminf(ax2, bx2), ry = fminf(ay2, by2);
          float iwd = fmaxf(rx - lx, 0.f), ihd = fmaxf(ry - ly, 0.f);
          float inter = iwd * ihd;
          float iou = inter / (area[i] + area[t] - inter + 1e-9f);
          if (iou > NMS_T) kp[t] = 0;
        }
      }
    }
    __syncthreads();
  }
  if (tid == 0) {
    int cnt = 0;
    for (int i = 0; i < 400 && cnt < DPI; ++i) {
      if (kp[i]) {
        int o = b * DPI + cnt;
        outB[o * 4 + 0] = x1[i]; outB[o * 4 + 1] = y1[i];
        outB[o * 4 + 2] = x2[i]; outB[o * 4 + 3] = y2[i];
        outS[o] = sc[i];
        outL[o] = (float)lb[i];
        ++cnt;
      }
    }
    for (; cnt < DPI; ++cnt) {
      int o = b * DPI + cnt;
      outB[o * 4 + 0] = 0.f; outB[o * 4 + 1] = 0.f;
      outB[o * 4 + 2] = 0.f; outB[o * 4 + 3] = 0.f;
      outS[o] = 0.f;
      outL[o] = -1.f;
    }
  }
}

// ---------------------------------------------------------------------------
extern "C" void kernel_launch(void* const* d_in, const int* in_sizes, int n_in,
                              void* d_out, int out_size, void* d_ws, size_t ws_size,
                              hipStream_t stream)
{
  (void)in_sizes; (void)n_in; (void)out_size; (void)ws_size;
  const float* f[4] = {(const float*)d_in[0], (const float*)d_in[1],
                       (const float*)d_in[2], (const float*)d_in[3]};
  const float* clsTW = (const float*)d_in[4];
  const float* clsTB = (const float*)d_in[5];
  const float* clsOW = (const float*)d_in[6];
  const float* clsOB = (const float*)d_in[7];
  const float* regTW = (const float*)d_in[8];
  const float* regTB = (const float*)d_in[9];
  const float* regOW = (const float*)d_in[10];
  const float* regOB = (const float*)d_in[11];
  const int* imgH = (const int*)d_in[12];
  const int* imgW = (const int*)d_in[13];

  const int HWs[4] = {4096, 1024, 256, 64};
  const int Hs[4]  = {64, 32, 16, 8};
  const size_t cumHW[4] = {0, 4096, 5120, 5376};

  char* ws = (char*)d_ws;
  size_t off = 0;
  auto alloc = [&](size_t bytes) -> char* {
    char* pp = ws + off;
    off += (bytes + 255) & ~(size_t)255;
    return pp;
  };

  _Float16* x16[4];
  for (int l = 0; l < 4; ++l)
    x16[l] = (_Float16*)alloc((size_t)BATCH * HWs[l] * 256 * 2);
  _Float16* ping  = (_Float16*)alloc((size_t)BATCH * 4096 * 256 * 2);
  _Float16* pong  = (_Float16*)alloc((size_t)BATCH * 4096 * 256 * 2);
  _Float16* wClsT = (_Float16*)alloc((size_t)1024 * 2304 * 2);
  _Float16* wRegT = (_Float16*)alloc((size_t)1024 * 2304 * 2);
  _Float16* wClsO = (_Float16*)alloc((size_t)720 * 2304 * 2);
  _Float16* wRegO = (_Float16*)alloc((size_t)36 * 2304 * 2);
  _Float16* zbuf  = (_Float16*)alloc((size_t)512 * 2);
  float* clsBuf = (float*)alloc((size_t)BATCH * 5440 * 720 * 4);
  float* regBuf = (float*)alloc((size_t)BATCH * 5440 * 36 * 4);
  float* topS  = (float*)alloc((size_t)BATCH * 4 * DPI * 4);
  int*   topI  = (int*)  alloc((size_t)BATCH * 4 * DPI * 4);
  float* candB = (float*)alloc((size_t)BATCH * 400 * 4 * 4);
  float* candS = (float*)alloc((size_t)BATCH * 400 * 4);
  int*   candL = (int*)  alloc((size_t)BATCH * 400 * 4);
  float* sortBv = (float*)alloc((size_t)BATCH * 400 * 4 * 4);
  float* sortSv = (float*)alloc((size_t)BATCH * 400 * 4);
  int*   sortLv = (int*)  alloc((size_t)BATCH * 400 * 4);

  // zero halo buffer + weight repack fp32 OIHW -> f16 [row][tap][cin]
  zero_f16<<<2, 256, 0, stream>>>(zbuf, 512);
  weights_to_f16<<<(1024 * 2304 + 255) / 256, 256, 0, stream>>>(clsTW, wClsT, 1024);
  weights_to_f16<<<(1024 * 2304 + 255) / 256, 256, 0, stream>>>(regTW, wRegT, 1024);
  weights_to_f16<<<(720 * 2304 + 255) / 256, 256, 0, stream>>>(clsOW, wClsO, 720);
  weights_to_f16<<<(36 * 2304 + 255) / 256, 256, 0, stream>>>(regOW, wRegO, 36);

  // activations fp32 NCHW -> f16 NHWC
  for (int l = 0; l < 4; ++l) {
    int total = BATCH * 256 * HWs[l];
    nchw_to_nhwc_f16<<<(total + 255) / 256, 256, 0, stream>>>(f[l], x16[l], 256,
                                                              HWs[l], total);
  }

  // towers + heads
  for (int head = 0; head < 2; ++head) {
    const _Float16* wT = head ? wRegT : wClsT;
    const float*    bT = head ? regTB : clsTB;
    const _Float16* wO = head ? wRegO : wClsO;
    const float*    bO = head ? regOB : clsOB;
    const int CoutO = head ? 36 : 720;
    float* outBuf = head ? regBuf : clsBuf;
    for (int l = 0; l < 4; ++l) {
      const _Float16* cur = x16[l];
      for (int layer = 0; layer < 4; ++layer) {
        _Float16* dst = (layer & 1) ? pong : ping;
        conv3x3_wmma<<<dim3(HWs[l] / 64, 2, BATCH), 256, 0, stream>>>(
            cur, wT + (size_t)layer * 256 * 2304, bT + layer * 256, zbuf,
            dst, nullptr, Hs[l], Hs[l], 256, 1);
        cur = dst;
      }
      float* region = outBuf + (size_t)CoutO * BATCH * cumHW[l];
      conv3x3_wmma<<<dim3(HWs[l] / 64, (CoutO + 127) / 128, BATCH), 256, 0, stream>>>(
          cur, wO, bO, zbuf, nullptr, region, Hs[l], Hs[l], CoutO, 0);
    }
  }

  // post-processing
  sigmoid_mask<<<2048, 256, 0, stream>>>(clsBuf, (size_t)BATCH * 5440 * 720);
  topk100<<<dim3(4, BATCH), 256, 0, stream>>>(clsBuf, topS, topI);
  decode_boxes<<<(BATCH * 4 * DPI + 255) / 256, 256, 0, stream>>>(
      topS, topI, regBuf, imgH, imgW, candB, candS, candL);
  sort_cands<<<BATCH, 256, 0, stream>>>(candB, candS, candL, sortBv, sortSv, sortLv);

  float* outB = (float*)d_out;
  float* outS = outB + BATCH * DPI * 4;
  float* outL = outS + BATCH * DPI;
  nms_final<<<BATCH, 256, 0, stream>>>(sortBv, sortSv, sortLv, imgH, imgW,
                                       outB, outS, outL);
}
